// NonLocalBlock_55241869361701
// MI455X (gfx1250) — matrix operand order repaired and checked by reading, not compiled, not run on hardware
//
#include <hip/hip_runtime.h>
#include <hip/hip_bf16.h>

#define LDIM  6272
#define CDIM  256
#define CHDIM 128
#define NB    2
#define NTILES (LDIM / 16)   // 392

typedef __attribute__((ext_vector_type(16))) _Float16 v16h;
typedef __attribute__((ext_vector_type(8)))  _Float16 v8h;
typedef __attribute__((ext_vector_type(8)))  float    v8f;

union V16U { v16h v; v8h h[2]; };

// A-fragment (16x32 f16): lane l<16 -> row M=l, K = kb+{0..7, 16..23};
//                         lane l>=16 -> row M=l-16, K = kb+{8..15, 24..31}
__device__ __forceinline__ v16h load_a_frag(const _Float16* __restrict__ row, int kb, int hl) {
  V16U u;
  u.h[0] = *(const v8h*)(row + kb + 8 * hl);
  u.h[1] = *(const v8h*)(row + kb + 16 + 8 * hl);
  return u.v;
}

// B-fragment (32x16 f16): lane n holds column n; lanes 0-15: K = kb+{0..7, 16..23},
// lanes 16-31: K = kb+{8..15, 24..31}.
__device__ __forceinline__ v16h load_b_frag(const _Float16* __restrict__ row, int kb, int hl) {
  V16U u;
  u.h[0] = *(const v8h*)(row + kb + 8 * hl);
  u.h[1] = *(const v8h*)(row + kb + 16 + 8 * hl);
  return u.v;
}

__device__ __forceinline__ v8f wmma_f16(v16h a, v16h b, v8f c) {
  return __builtin_amdgcn_wmma_f32_16x16x32_f16(false, a, false, b, (short)0, c, false, false);
}

// Build f16 A-fragment from an fp32 row (contiguous in K).
__device__ __forceinline__ v16h make_a_from_f32(const float* __restrict__ wrow, int kb, int hl) {
  v16h a;
#pragma unroll
  for (int p = 0; p < 8; ++p) {
    a[p]     = (_Float16)wrow[kb + 8 * hl + p];
    a[8 + p] = (_Float16)wrow[kb + 16 + 8 * hl + p];
  }
  return a;
}

// ---------------------------------------------------------------------------
// One projection (Ch x L tile of 16 columns), templated on output layout so
// every call keeps a direct kernarg->pointer chain (global_load/store, not
// flat).  TMAJOR=true: dst is (n, L, 128) f16 (theta^T / phi^T).
// TMAJOR=false: dst is (n, 128, L) f16 (g).
// ---------------------------------------------------------------------------
template <bool TMAJOR>
__device__ __forceinline__ void do_proj(const v16h (&bx)[8],
                                        const float* __restrict__ W,
                                        const float* __restrict__ Bv,
                                        _Float16* __restrict__ dst,
                                        int n, int i0, int hl, int ln) {
#pragma unroll
  for (int ht = 0; ht < 8; ++ht) {
    const int h0 = ht * 16;
    const float* wrow = W + (size_t)(h0 + ln) * CDIM;  // A row per lane
    v8f acc = {};
#pragma unroll
    for (int k = 0; k < 8; ++k)
      acc = wmma_f16(make_a_from_f32(wrow, 32 * k, hl), bx[k], acc);
    // D layout: row h = h0 + v + 8*hl, col i = i0 + ln
#pragma unroll
    for (int v = 0; v < 8; ++v) {
      const int h = h0 + v + 8 * hl;
      const int i = i0 + ln;
      const _Float16 hv = (_Float16)(acc[v] + Bv[h]);
      if (TMAJOR) dst[((size_t)n * LDIM + i) * CHDIM + h] = hv;
      else        dst[((size_t)n * CHDIM + h) * LDIM + i] = hv;
    }
  }
}

// ---------------------------------------------------------------------------
// Kernel 1: projections.  One wave per 16-column tile of L.
// ---------------------------------------------------------------------------
__global__ void k_proj(const float* __restrict__ x,
                       const float* __restrict__ gw, const float* __restrict__ gb,
                       const float* __restrict__ tw, const float* __restrict__ tb,
                       const float* __restrict__ pw, const float* __restrict__ pb,
                       _Float16* __restrict__ gm,
                       _Float16* __restrict__ thT,
                       _Float16* __restrict__ phT) {
  const int lane = threadIdx.x & 31;
  const int wid  = blockIdx.x * (blockDim.x >> 5) + (threadIdx.x >> 5);
  const int n  = wid / NTILES;
  const int i0 = (wid % NTILES) * 16;
  const int hl = lane >> 4, ln = lane & 15;

  // Preload B_x fragments for all 8 K-chunks (K = C = 256).
  // B[K=c-kb, n=i]: lane ln -> column i0+ln; per-lane K's are c = kb+8*hl+p+(p&8).
  v16h bx[8];
#pragma unroll
  for (int k = 0; k < 8; ++k) {
    const float* base = x + ((size_t)n * CDIM + 32 * k + 8 * hl) * LDIM + i0 + ln;
    v16h t;
#pragma unroll
    for (int p = 0; p < 16; ++p) t[p] = (_Float16)base[(size_t)(p + (p & 8)) * LDIM];
    bx[k] = t;
  }

  do_proj<true >(bx, tw, tb, thT, n, i0, hl, ln);
  do_proj<true >(bx, pw, pb, phT, n, i0, hl, ln);
  do_proj<false>(bx, gw, gb, gm,  n, i0, hl, ln);
}

// ---------------------------------------------------------------------------
// Kernel 2: flash-attention core.  One wave owns a 16-query tile; theta
// B-fragments stay resident in VGPRs; stream over j in steps of 32.
// Scores computed transposed (S'[j,i] = phi^T theta) so exp'd P converts
// to the PV A-fragment fully in-lane.
// ---------------------------------------------------------------------------
__global__ void k_attn(const _Float16* __restrict__ thT,
                       const _Float16* __restrict__ phT,
                       const _Float16* __restrict__ gm,
                       _Float16* __restrict__ yT) {
  const int lane = threadIdx.x & 31;
  const int wid  = blockIdx.x * (blockDim.x >> 5) + (threadIdx.x >> 5);
  const int n  = wid / NTILES;
  const int i0 = (wid % NTILES) * 16;
  const int hl = lane >> 4, ln = lane & 15;

  // Resident B_theta (K = Ch = 128 -> 4 chunks): lane ln -> query column i0+ln.
  const _Float16* throw_ = thT + ((size_t)n * LDIM + i0 + ln) * CHDIM;
  v16h bth[4];
#pragma unroll
  for (int k = 0; k < 4; ++k) bth[k] = load_b_frag(throw_, 32 * k, hl);

  v8f acc[8];
#pragma unroll
  for (int cc = 0; cc < 8; ++cc) acc[cc] = (v8f){};

  float m_run = -1e30f, l_run = 0.0f;

  const _Float16* phbase = phT + (size_t)n * LDIM * CHDIM;
  const _Float16* gbase  = gm  + (size_t)n * CHDIM * LDIM;

  for (int j0 = 0; j0 < LDIM; j0 += 32) {
    // ---- prefetch next j-tile (lowers to global_prefetch_b8) --------------
    if (j0 + 32 < LDIM) {
      __builtin_prefetch(phbase + (size_t)(j0 + 32 + ln) * CHDIM, 0, 3);
      __builtin_prefetch(phbase + (size_t)(j0 + 48 + ln) * CHDIM, 0, 3);
      __builtin_prefetch(gbase + (size_t)(lane * 4) * LDIM + j0 + 32, 0, 3);
    }

    // ---- scores: S'a rows j0..j0+15, S'b rows j0+16..j0+31 (transposed) ----
    const _Float16* pha = phbase + (size_t)(j0 + ln) * CHDIM;
    const _Float16* phb = phbase + (size_t)(j0 + 16 + ln) * CHDIM;
    v8f sa = {}, sb = {};
#pragma unroll
    for (int k = 0; k < 4; ++k) {
      sa = wmma_f16(load_a_frag(pha, 32 * k, hl), bth[k], sa);
      sb = wmma_f16(load_a_frag(phb, 32 * k, hl), bth[k], sb);
    }

    // ---- online softmax over j (this lane sees 16 of 32 j's for col i=ln) --
    float mloc = sa[0];
#pragma unroll
    for (int v = 1; v < 8; ++v) mloc = fmaxf(mloc, sa[v]);
#pragma unroll
    for (int v = 0; v < 8; ++v) mloc = fmaxf(mloc, sb[v]);
    mloc = fmaxf(mloc, __shfl_xor(mloc, 16, 32));      // combine lane halves
    const float m_new = fmaxf(m_run, mloc);
    const float scale = __expf(m_run - m_new);
    m_run = m_new;

    float ssum = 0.0f;
    v8f pa, pb;
#pragma unroll
    for (int v = 0; v < 8; ++v) {
      pa[v] = __expf(sa[v] - m_run);
      pb[v] = __expf(sb[v] - m_run);
      ssum += pa[v] + pb[v];
    }
    ssum += __shfl_xor(ssum, 16, 32);
    l_run = l_run * scale + ssum;

    // ---- rescale y accumulator (rows are VGPR/half-lane indexed) ----------
    float srow[8];
#pragma unroll
    for (int v = 0; v < 8; ++v) srow[v] = __shfl(scale, v + 8 * hl, 32);
#pragma unroll
    for (int cc = 0; cc < 8; ++cc)
#pragma unroll
      for (int v = 0; v < 8; ++v) acc[cc][v] *= srow[v];

    // ---- pack P' -> A fragment (16 x 32, K = j), purely in-lane -----------
    v16h ap;
#pragma unroll
    for (int p = 0; p < 8; ++p) {
      ap[p]     = (_Float16)pa[p];   // K = 8*hl + p      (tile a)
      ap[8 + p] = (_Float16)pb[p];   // K = 16 + 8*hl + p (tile b)
    }

    // ---- y[i, c] += P[i, j] * g[c, j] : 8 channel chunks of 16 ------------
#pragma unroll
    for (int cc = 0; cc < 8; ++cc) {
      const _Float16* grow = gbase + (size_t)(cc * 16 + ln) * LDIM + j0;
      acc[cc] = wmma_f16(ap, load_b_frag(grow, 0, hl), acc[cc]);
    }
  }

  // ---- normalize and store y^T (n, L, 128) f16 ----------------------------
  const float inv = 1.0f / l_run;
  float srow[8];
#pragma unroll
  for (int v = 0; v < 8; ++v) srow[v] = __shfl(inv, v + 8 * hl, 32);
#pragma unroll
  for (int cc = 0; cc < 8; ++cc)
#pragma unroll
    for (int v = 0; v < 8; ++v) {
      const int i = i0 + v + 8 * hl;
      yT[((size_t)n * LDIM + i) * CHDIM + cc * 16 + ln] =
          (_Float16)(acc[cc][v] * srow[v]);
    }
}

// ---------------------------------------------------------------------------
// Kernel 3: z = wz * y + b + x (residual), fp32 output.
// ---------------------------------------------------------------------------
__global__ void k_out(const _Float16* __restrict__ yT,
                      const float* __restrict__ wz,
                      const float* __restrict__ wzb,
                      const float* __restrict__ x,
                      float* __restrict__ out) {
  const int lane = threadIdx.x & 31;
  const int wid  = blockIdx.x * (blockDim.x >> 5) + (threadIdx.x >> 5);
  const int n  = wid / NTILES;
  const int i0 = (wid % NTILES) * 16;
  const int hl = lane >> 4, ln = lane & 15;

  // Resident B_y (K = Ch = 128 -> 4 chunks).
  const _Float16* yrow = yT + ((size_t)n * LDIM + i0 + ln) * CHDIM;
  v16h by[4];
#pragma unroll
  for (int k = 0; k < 4; ++k) by[k] = load_b_frag(yrow, 32 * k, hl);

#pragma unroll 4
  for (int ot = 0; ot < 16; ++ot) {
    const int o0 = ot * 16;
    const float* wrow = wz + (size_t)(o0 + ln) * CHDIM;
    v8f acc = {};
#pragma unroll
    for (int k = 0; k < 4; ++k)
      acc = wmma_f16(make_a_from_f32(wrow, 32 * k, hl), by[k], acc);
#pragma unroll
    for (int v = 0; v < 8; ++v) {
      const int o = o0 + v + 8 * hl;
      const size_t idx = ((size_t)n * CDIM + o) * LDIM + i0 + ln;
      out[idx] = acc[v] + wzb[o] + x[idx];
    }
  }
}

// ---------------------------------------------------------------------------
extern "C" void kernel_launch(void* const* d_in, const int* in_sizes, int n_in,
                              void* d_out, int out_size, void* d_ws, size_t ws_size,
                              hipStream_t stream) {
  const float* x   = (const float*)d_in[0];
  const float* gw  = (const float*)d_in[1];
  const float* gb  = (const float*)d_in[2];
  const float* tw  = (const float*)d_in[3];
  const float* tb  = (const float*)d_in[4];
  const float* pw  = (const float*)d_in[5];
  const float* pb  = (const float*)d_in[6];
  const float* wzw = (const float*)d_in[7];
  const float* wzb = (const float*)d_in[8];
  float* out = (float*)d_out;

  const size_t hp = (size_t)NB * LDIM * CHDIM;     // halves per array
  _Float16* thT = (_Float16*)d_ws;
  _Float16* phT = thT + hp;
  _Float16* gm  = phT + hp;
  _Float16* yT  = gm + hp;                          // total ~12.3 MB of d_ws

  const dim3 block(128);                            // 4 waves (wave32)
  const dim3 grid(NB * NTILES / 4);                 // 196 blocks, exact

  k_proj<<<grid, block, 0, stream>>>(x, gw, gb, tw, tb, pw, pb, gm, thT, phT);
  k_attn<<<grid, block, 0, stream>>>(thT, phT, gm, yT);
  k_out <<<grid, block, 0, stream>>>(yT, wzw, wzb, x, out);
}
